// ScaleDotProductAttention_15676630630672
// MI455X (gfx1250) — matrix operand order repaired and checked by reading, not compiled, not run on hardware
//
#include <hip/hip_runtime.h>
#include <hip/hip_bf16.h>

// ---------------------------------------------------------------------------
// Fused SDPA with materialized attention matrix, bf16 WMMA path for gfx1250.
//   d_out = [ context (B*S*D fp32) | attn (B*S*S fp32) ]
// One WG = (batch, 32 query rows); 8 wave32. Scores live in LDS (256KB of the
// 320KB WGP pool) so softmax is exact two-pass and P*V reads straight from LDS.
// ---------------------------------------------------------------------------

typedef __attribute__((ext_vector_type(16))) __bf16 v16bf;
typedef __attribute__((ext_vector_type(8)))  float  v8f;

#define B_   16
#define S_   2048
#define D_   128
#define TQ   32          // query rows per workgroup
#define NWAVE 8
#define NTHREADS 256     // 8 wave32
#define KTILES (S_/16)   // 128 key tiles of 16

// LDS layout (dynamic)
#define SC_OFF   0                       // 32*2048 fp32 scores/probs  = 262144
#define QT_OFF   262144                  // 32*128 bf16 Q tile         =   8192
#define RED_OFF  270336                  // 32*8 fp32                  =   1024
#define RM_OFF   271360                  // 32 fp32                    =    128
#define RS_OFF   271488                  // 32 fp32                    =    128
#define SMEM_BYTES 271616

// fp32 -> bf16 round-to-nearest-even
__device__ __forceinline__ unsigned short f2bf(float x) {
    unsigned int u = __float_as_uint(x);
    unsigned int r = u + 0x7FFFu + ((u >> 16) & 1u);
    return (unsigned short)(r >> 16);
}

// ---------------------------------------------------------------------------
// Prep: K -> bf16 row-major, V -> bf16 transposed Vt[b][d][s] so the P*V
// B-fragment (fixed output column across 16 consecutive key rows) is a
// contiguous 32B per-lane load.
// ---------------------------------------------------------------------------
__global__ void prep_kv(const float* __restrict__ K, const float* __restrict__ V,
                        unsigned short* __restrict__ Kb, unsigned short* __restrict__ Vt) {
    size_t i = (size_t)blockIdx.x * blockDim.x + threadIdx.x;
    const size_t total = (size_t)B_ * S_ * D_;
    if (i >= total) return;
    int d = (int)(i % D_);
    int s = (int)((i / D_) % S_);
    int b = (int)(i / ((size_t)S_ * D_));
    Kb[i] = f2bf(K[i]);
    Vt[((size_t)b * D_ + d) * (size_t)S_ + s] = f2bf(V[i]);
}

// ---------------------------------------------------------------------------
__global__ __launch_bounds__(NTHREADS, 1)
void attn_kernel(const float* __restrict__ Q, const unsigned char* __restrict__ mask,
                 const unsigned short* __restrict__ Kb, const unsigned short* __restrict__ Vt,
                 float* __restrict__ ctx_out, float* __restrict__ attn_out) {
    extern __shared__ char smem[];
    float*          sc     = (float*)(smem + SC_OFF);   // [32][2048]
    unsigned short* qt     = (unsigned short*)(smem + QT_OFF);
    float*          red    = (float*)(smem + RED_OFF);
    float*          rowm   = (float*)(smem + RM_OFF);
    float*          rowsum = (float*)(smem + RS_OFF);

    const int tid  = threadIdx.x;
    // force scalar (uniform) wave id -> scalar loop control, EXEC stays all-1s
    const int wave = __builtin_amdgcn_readfirstlane(tid >> 5);
    const int lane = tid & 31;
    const int m    = lane & 15;       // WMMA fragment row/column index
    const int half = lane >> 4;       // half-wave selector

    const int b  = blockIdx.x / (S_ / TQ);
    const int q0 = (blockIdx.x % (S_ / TQ)) * TQ;

    // ---- stage Q tile (32x128) as bf16 into LDS -------------------------
    const float* Qt = Q + ((size_t)b * S_ + q0) * D_;
    for (int i = tid; i < TQ * D_; i += NTHREADS)
        qt[i] = f2bf(Qt[i]);
    __syncthreads();

    // ---- build Q A-fragments: 2 query halves x 4 d-chunks ---------------
    // contiguous k order used here (B matches): VGPR pair v holds K = half*16 + v*2 + {0,1}
    v16bf afrag[2][4];
    #pragma unroll
    for (int qh = 0; qh < 2; ++qh) {
        #pragma unroll
        for (int dc = 0; dc < 4; ++dc) {
            union { v16bf v; unsigned int u[8]; } a;
            #pragma unroll
            for (int v = 0; v < 8; ++v) {
                int k = dc * 32 + v * 2 + half * 16;
                a.u[v] = *(const unsigned int*)&qt[(qh * 16 + m) * D_ + k];
            }
            afrag[qh][dc] = a.v;
        }
    }

    const float scale = 0.08838834764831845f;                 // 1/sqrt(128)
    const unsigned char* mrow = mask + (size_t)b * S_ * S_;   // q=0 row (broadcast)

    // =====================================================================
    // Pass 1: scores = Q K^T * scale, masked, into LDS.
    // Each K-fragment feeds both query halves (2x WMMA per load).
    // =====================================================================
    for (int t = wave; t < KTILES; t += NWAVE) {
        const int kcol = t * 16;
        const unsigned short* kbase =
            Kb + ((size_t)(b * S_ + kcol + m) * D_) + half * 16;
        __builtin_prefetch(kbase + (size_t)NWAVE * 16 * D_, 0, 0);

        // load all 4 B fragments (32x16 bf16 each): 8x 16B per lane
        union { v16bf v; unsigned int u[8]; } bfr[4];
        #pragma unroll
        for (int dc = 0; dc < 4; ++dc) {
            const uint4* kp4 = (const uint4*)(kbase + dc * 32);
            uint4 lo = kp4[0];
            uint4 hi = kp4[1];
            bfr[dc].u[0] = lo.x; bfr[dc].u[1] = lo.y; bfr[dc].u[2] = lo.z; bfr[dc].u[3] = lo.w;
            bfr[dc].u[4] = hi.x; bfr[dc].u[5] = hi.y; bfr[dc].u[6] = hi.z; bfr[dc].u[7] = hi.w;
        }

        v8f acc0 = {}, acc1 = {};
        #pragma unroll
        for (int dc = 0; dc < 4; ++dc) {
            acc0 = __builtin_amdgcn_wmma_f32_16x16x32_bf16(
                false, afrag[0][dc], false, bfr[dc].v, (short)0, acc0, false, false);
            acc1 = __builtin_amdgcn_wmma_f32_16x16x32_bf16(
                false, afrag[1][dc], false, bfr[dc].v, (short)0, acc1, false, false);
        }

        const bool msk = (mrow[kcol + m] != 0);
        #pragma unroll
        for (int v = 0; v < 8; ++v) {
            float s0 = msk ? -__builtin_inff() : acc0[v] * scale;
            float s1 = msk ? -__builtin_inff() : acc1[v] * scale;
            sc[(half * 8 + v) * S_ + kcol + m]        = s0;
            sc[(16 + half * 8 + v) * S_ + kcol + m]   = s1;
        }
    }
    __syncthreads();

    // =====================================================================
    // Softmax: 32 rows, 8 threads/row, interleaved float4 chunks
    // =====================================================================
    {
        const int r    = tid >> 3;
        const int lsub = tid & 7;
        const int rb   = r * S_;

        float mx = -__builtin_inff();
        for (int j = 0; j < 64; ++j) {
            float4 s4 = *(const float4*)&sc[rb + lsub * 4 + j * 32];
            mx = fmaxf(mx, fmaxf(fmaxf(s4.x, s4.y), fmaxf(s4.z, s4.w)));
        }
        red[r * 8 + lsub] = mx;
        __syncthreads();
        if (tid < TQ) {
            float m2 = red[tid * 8];
            #pragma unroll
            for (int j = 1; j < 8; ++j) m2 = fmaxf(m2, red[tid * 8 + j]);
            rowm[tid] = m2;
        }
        __syncthreads();

        const float rm = rowm[r];
        float sum = 0.f;
        for (int j = 0; j < 64; ++j) {
            float4* p4 = (float4*)&sc[rb + lsub * 4 + j * 32];
            float4 s4 = *p4;
            s4.x = __expf(s4.x - rm);
            s4.y = __expf(s4.y - rm);
            s4.z = __expf(s4.z - rm);
            s4.w = __expf(s4.w - rm);
            *p4 = s4;
            sum += s4.x + s4.y + s4.z + s4.w;
        }
        red[r * 8 + lsub] = sum;
        __syncthreads();
        if (tid < TQ) {
            float s2 = 0.f;
            #pragma unroll
            for (int j = 0; j < 8; ++j) s2 += red[tid * 8 + j];
            rowsum[tid] = s2;
        }
        __syncthreads();

        const float inv = 1.0f / rowsum[r];
        float4* dst = (float4*)(attn_out + ((size_t)b * S_ + q0 + r) * S_);
        for (int j = 0; j < 64; ++j) {
            float4* p4 = (float4*)&sc[rb + lsub * 4 + j * 32];
            float4 s4 = *p4;
            s4.x *= inv; s4.y *= inv; s4.z *= inv; s4.w *= inv;
            *p4 = s4;                               // keep normalized P in LDS
            dst[lsub + j * 8] = s4;                 // coalesced attn write
        }
    }
    __syncthreads();

    // =====================================================================
    // Pass 2: context = P (LDS fp32 -> bf16) x V (Vt bf16).
    // waves 0-3: query half 0, waves 4-7: query half 1; each wave owns two
    // 16-wide d-tiles and shares one A fragment per kk across both.
    // =====================================================================
    {
        const int qh  = __builtin_amdgcn_readfirstlane(wave >> 2);
        const int dt0 = __builtin_amdgcn_readfirstlane((wave & 3) * 2);

        const float* prow = sc + (qh * 16 + m) * S_;
        const unsigned short* vb0 =
            Vt + ((size_t)b * D_ + (dt0 + 0) * 16 + m) * (size_t)S_ + half * 16;
        const unsigned short* vb1 =
            Vt + ((size_t)b * D_ + (dt0 + 1) * 16 + m) * (size_t)S_ + half * 16;

        v8f acc0 = {}, acc1 = {};
        for (int kk = 0; kk < S_ / 32; ++kk) {
            // A fragment from probabilities in LDS (fp32 pairs -> packed bf16)
            union { v16bf v; unsigned int u[8]; } a;
            #pragma unroll
            for (int v = 0; v < 8; ++v) {
                int k = kk * 32 + v * 2 + half * 16;
                float2 pv = *(const float2*)&prow[k];
                a.u[v] = (unsigned int)f2bf(pv.x) | ((unsigned int)f2bf(pv.y) << 16);
            }
            // B fragments: 16 contiguous bf16 key rows per lane, both d-tiles
            union { v16bf v; unsigned int u[8]; } b0, b1;
            {
                const uint4* p = (const uint4*)(vb0 + kk * 32);
                uint4 lo = p[0], hi = p[1];
                b0.u[0] = lo.x; b0.u[1] = lo.y; b0.u[2] = lo.z; b0.u[3] = lo.w;
                b0.u[4] = hi.x; b0.u[5] = hi.y; b0.u[6] = hi.z; b0.u[7] = hi.w;
            }
            {
                const uint4* p = (const uint4*)(vb1 + kk * 32);
                uint4 lo = p[0], hi = p[1];
                b1.u[0] = lo.x; b1.u[1] = lo.y; b1.u[2] = lo.z; b1.u[3] = lo.w;
                b1.u[4] = hi.x; b1.u[5] = hi.y; b1.u[6] = hi.z; b1.u[7] = hi.w;
            }
            acc0 = __builtin_amdgcn_wmma_f32_16x16x32_bf16(
                false, a.v, false, b0.v, (short)0, acc0, false, false);
            acc1 = __builtin_amdgcn_wmma_f32_16x16x32_bf16(
                false, a.v, false, b1.v, (short)0, acc1, false, false);
        }
        #pragma unroll
        for (int v = 0; v < 8; ++v) {
            const int row = qh * 16 + half * 8 + v;
            float* crow = ctx_out + ((size_t)b * S_ + q0 + row) * D_;
            crow[(dt0 + 0) * 16 + m] = acc0[v];
            crow[(dt0 + 1) * 16 + m] = acc1[v];
        }
    }
}

// ---------------------------------------------------------------------------
extern "C" void kernel_launch(void* const* d_in, const int* in_sizes, int n_in,
                              void* d_out, int out_size, void* d_ws, size_t ws_size,
                              hipStream_t stream) {
    const float*         Q    = (const float*)d_in[0];
    const float*         K    = (const float*)d_in[1];
    const float*         V    = (const float*)d_in[2];
    const unsigned char* mask = (const unsigned char*)d_in[3];

    float* ctx  = (float*)d_out;                       // B*S*D fp32
    float* attn = ctx + (size_t)B_ * S_ * D_;          // B*S*S fp32

    unsigned short* Kb = (unsigned short*)d_ws;                    // 8 MB
    unsigned short* Vt = Kb + (size_t)B_ * S_ * D_;                // 8 MB

    const int total = B_ * S_ * D_;
    prep_kv<<<(total + 255) / 256, 256, 0, stream>>>(K, V, Kb, Vt);

    attn_kernel<<<B_ * (S_ / TQ), NTHREADS, SMEM_BYTES, stream>>>(
        Q, mask, Kb, Vt, ctx, attn);
}
